// DynamicSparseAttention_13932873908663
// MI455X (gfx1250) — matrix operand order repaired and checked by reading, not compiled, not run on hardware
//
#include <hip/hip_runtime.h>
#include <stdint.h>
#include <math.h>

#define B_    2
#define S_    2048
#define NH_   16
#define HD_   64
#define HID_  1024
#define M_    (B_*S_)      // 4096 rows of hidden_states
#define WIN_  1024
#define TOPK_ 1024
#define STRIDE_ 45         // int(isqrt(2048))

typedef __attribute__((ext_vector_type(16))) __bf16 v16bf;
typedef __attribute__((ext_vector_type(8)))  float  v8f;

union TileU {
  uint4    q[2];
  uint16_t u[16];
  v16bf    v;
};

static __device__ __forceinline__ uint16_t f2bf(float f) {
  union { float f; uint32_t u; } x; x.f = f;
  uint32_t u = x.u;
  if ((u & 0x7F800000u) == 0x7F800000u) return (uint16_t)(u >> 16); // inf/nan passthrough
  u += 0x7FFFu + ((u >> 16) & 1u);                                  // round-nearest-even
  return (uint16_t)(u >> 16);
}

// Order-preserving float->u32 key (ascending) and inverse.
static __device__ __forceinline__ uint32_t fkey(float f) {
  union { float f; uint32_t u; } x; x.f = f;
  return (x.u & 0x80000000u) ? ~x.u : (x.u | 0x80000000u);
}
static __device__ __forceinline__ float keyf(uint32_t k) {
  union { float f; uint32_t u; } x;
  x.u = (k & 0x80000000u) ? (k ^ 0x80000000u) : ~k;
  return x.f;
}

// A operand: 16x32 bf16 tile, row-major source, tile = &A[m0][k0], leading dim ld.
// Lane L (0..15): M=L, K = {0..7, 16..23}; lane L+16: M=L, K = {8..15, 24..31}.
static __device__ __forceinline__ v16bf ldA(const uint16_t* tile, int ld) {
  int lane = threadIdx.x & 31;
  const uint16_t* r = tile + (size_t)(lane & 15) * ld + ((lane >> 4) << 3);
  TileU x;
  x.q[0] = *(const uint4*)(r);
  x.q[1] = *(const uint4*)(r + 16);
  return x.v;
}

// B operand: 32x16 bf16 tile (KxN), row-major source, tile = &B[k0][n0], leading dim ld.
// Lane L (0..15): N=L, K = {0..7, 16..23}; lane L+16: N=L, K = {8..15, 24..31}.
static __device__ __forceinline__ v16bf ldB(const uint16_t* tile, int ld) {
  int lane = threadIdx.x & 31;
  const uint16_t* r = tile + (lane & 15) + (size_t)((lane >> 4) << 3) * ld;
  TileU x;
#pragma unroll
  for (int s = 0; s < 8; s++) {
    x.u[s] = r[(size_t)s * ld];
    x.u[s + 8] = r[(size_t)(s + 16) * ld];
  }
  return x.v;
}

__global__ void k_init(float* meanq) {
  if (threadIdx.x == 0) meanq[0] = 0.f;
}

// Convert 4 weight matrices (transposed: Wt[k][n] = W[n][k]) and hidden states to bf16.
__global__ void k_convert(const float* wq, const float* wk, const float* wv, const float* wo,
                          const float* hs, uint16_t* wt, uint16_t* hsb) {
  long id = (long)blockIdx.x * blockDim.x + threadIdx.x;
  const long WTOT = 4L * HID_ * HID_;
  if (id < WTOT) {
    int w = (int)(id >> 20);              // HID_*HID_ == 1<<20
    int e = (int)(id & 0xFFFFF);
    int k = e >> 10, n = e & 1023;
    const float* src = (w == 0) ? wq : (w == 1) ? wk : (w == 2) ? wv : wo;
    wt[id] = f2bf(src[(size_t)n * HID_ + k]);
  } else {
    long e = id - WTOT;
    if (e < (long)M_ * HID_) hsb[e] = f2bf(hs[e]);
  }
}

// QKV projections: out = hs @ W^T  (bf16 WMMA, fp32 accumulate).
// Each wave computes a 16x64 output block: 1 A load feeds 4 WMMAs per K-step.
// z=0 -> Q[m][n] (+ mean(Q) accumulation), z=1 -> K transposed [b,h,d,s], z=2 -> V[m][n].
__global__ void k_proj_qkv(const uint16_t* __restrict__ hsb, const uint16_t* __restrict__ wt,
                           uint16_t* qb, uint16_t* ktr, uint16_t* vb, float* meanq) {
  int wave = threadIdx.x >> 5;
  int lane = threadIdx.x & 31;
  int n0 = (blockIdx.x * 4 + wave) * 64;  // 16 n-blocks of 64
  int m0 = blockIdx.y * 16;               // 256 m-tiles
  int z  = blockIdx.z;                    // 0=q 1=k 2=v
  const uint16_t* W = wt + (size_t)z * HID_ * HID_;
  const uint16_t* tileA = hsb + (size_t)m0 * HID_;
  v8f acc[4] = {};
  for (int kb = 0; kb < HID_; kb += 32) {
    if (kb + 64 < HID_) __builtin_prefetch((const void*)(tileA + kb + 64), 0, 0);
    v16bf a = ldA(tileA + kb, HID_);
#pragma unroll
    for (int t = 0; t < 4; t++) {
      v16bf b = ldB(W + (size_t)kb * HID_ + n0 + t * 16, HID_);
      acc[t] = __builtin_amdgcn_wmma_f32_16x16x32_bf16(false, a, false, b, (short)0, acc[t], false, false);
    }
  }
  int mo = (lane >> 4) * 8;
  if (z == 0) {
    float loc = 0.f;
#pragma unroll
    for (int t = 0; t < 4; t++) {
      int n = n0 + t * 16 + (lane & 15);
#pragma unroll
      for (int r = 0; r < 8; r++) {
        int m = m0 + r + mo;
        qb[(size_t)m * HID_ + n] = f2bf(acc[t][r]);
        loc += acc[t][r];
      }
    }
    for (int off = 16; off > 0; off >>= 1) loc += __shfl_down(loc, off, 32);
    if (lane == 0) atomicAdd(meanq, loc);
  } else if (z == 1) {
#pragma unroll
    for (int t = 0; t < 4; t++) {
      int n = n0 + t * 16 + (lane & 15);
      int h = n / HD_, d = n % HD_;
#pragma unroll
      for (int r = 0; r < 8; r++) {
        int m = m0 + r + mo;
        int bb = m / S_, s = m % S_;
        ktr[(size_t)(((bb * NH_ + h) * HD_) + d) * S_ + s] = f2bf(acc[t][r]);
      }
    }
  } else {
#pragma unroll
    for (int t = 0; t < 4; t++) {
      int n = n0 + t * 16 + (lane & 15);
#pragma unroll
      for (int r = 0; r < 8; r++) {
        int m = m0 + r + mo;
        vb[(size_t)m * HID_ + n] = f2bf(acc[t][r]);
      }
    }
  }
}

// Pattern predictor: argmax over 4 combined scores.
__global__ void k_pattern(const float* meanq, const float* slp_w, const float* slp_b,
                          const float* icp_w, const float* icp_b, const float* pw, int* pidx) {
  if (threadIdx.x != 0) return;
  float mq = meanq[0] / (float)((size_t)M_ * HID_);
  float lg = logf((float)(S_ + 1));
  float best = -INFINITY; int bi = 0;
  for (int p = 0; p < 4; p++) {
    float ssum = 0.f;
    for (int k = 0; k < HID_; k++) ssum += icp_w[p * HID_ + k];
    float comb = (slp_w[p] * lg + slp_b[p] + ssum * mq + icp_b[p]) * pw[p];
    if (comb > best) { best = comb; bi = p; }
  }
  pidx[0] = bi;
}

// scores = scale * Q Kt with pattern mask applied at store time; fp32 to ws.
// Each wave computes a 16x64 block of scores (4 WMMAs per K-step, 2 K-steps).
__global__ void k_scores(const uint16_t* __restrict__ qb, const uint16_t* __restrict__ ktr,
                         const int* __restrict__ pidx, float* __restrict__ sc, int b, int h) {
  int wave = threadIdx.x >> 5, lane = threadIdx.x & 31;
  int j0 = (blockIdx.x * 4 + wave) * 64;   // 32 j-blocks of 64
  int i0 = blockIdx.y * 16;                // 128 i-tiles
  const uint16_t* qt = qb  + (size_t)(b * S_ + i0) * HID_ + h * HD_;
  const uint16_t* kt = ktr + (size_t)((b * NH_ + h) * HD_) * S_ + j0;
  v8f acc[4] = {};
  for (int kb = 0; kb < HD_; kb += 32) {
    v16bf a = ldA(qt + kb, HID_);
#pragma unroll
    for (int t = 0; t < 4; t++) {
      v16bf bm = ldB(kt + (size_t)kb * S_ + t * 16, S_);
      acc[t] = __builtin_amdgcn_wmma_f32_16x16x32_bf16(false, a, false, bm, (short)0, acc[t], false, false);
    }
  }
  int idx = pidx[0];
  int mo = (lane >> 4) * 8;
#pragma unroll
  for (int t = 0; t < 4; t++) {
    int j = j0 + t * 16 + (lane & 15);
#pragma unroll
    for (int r = 0; r < 8; r++) {
      int i = i0 + r + mo;
      float v = acc[t][r] * 0.125f;   // HD^-0.5
      bool ok;
      if (idx == 0) {
        ok = (j >= i - WIN_ / 2) && (j < i + WIN_ / 2);
      } else if (idx == 1) {
        int d = i - j;
        int md = d % STRIDE_; if (md < 0) md += STRIDE_;   // Python modulo
        ok = (md == 0) || (d < STRIDE_ && d > -STRIDE_);
      } else if (idx == 2) {
        ok = (i / WIN_) == (j / WIN_);
      } else {
        ok = true;   // dynamic: top-k applied by selection stage (idempotent with sparsity step)
      }
      sc[(size_t)i * S_ + j] = ok ? v : -INFINITY;
    }
  }
}

// Per row: exact 1024-th largest via 4-pass radix select on LDS-staged row,
// then softmax with >= tv keep-mask; bf16 probs out.
__global__ void __launch_bounds__(256) k_select_softmax(const float* __restrict__ sc,
                                                        uint16_t* __restrict__ attn) {
  int row = blockIdx.x, tid = threadIdx.x;
  __shared__ float vals[S_];
  __shared__ unsigned hist[256];
  __shared__ unsigned s_prefix, s_rem, s_maxkey;
  __shared__ float s_denom;
  for (int j = tid; j < S_; j += 256) vals[j] = sc[(size_t)row * S_ + j];
  if (tid == 0) { s_prefix = 0u; s_rem = TOPK_; s_maxkey = 0u; s_denom = 0.f; }
  __syncthreads();
  unsigned lk = 0u;
  for (int j = tid; j < S_; j += 256) { unsigned k = fkey(vals[j]); if (k > lk) lk = k; }
  atomicMax(&s_maxkey, lk);
  for (int shift = 24; shift >= 0; shift -= 8) {
    hist[tid] = 0u;
    __syncthreads();
    unsigned himask = (shift == 24) ? 0u : (0xFFFFFFFFu << (shift + 8));
    unsigned pfx = s_prefix;
    for (int j = tid; j < S_; j += 256) {
      unsigned k = fkey(vals[j]);
      if ((k & himask) == (pfx & himask)) atomicAdd(&hist[(k >> shift) & 255u], 1u);
    }
    __syncthreads();
    if (tid == 0) {
      unsigned cum = 0u, rem = s_rem;
      for (int bin = 255; bin >= 0; bin--) {
        unsigned c = hist[bin];
        if (cum + c >= rem) { s_prefix |= ((unsigned)bin) << shift; s_rem = rem - cum; break; }
        cum += c;
      }
    }
    __syncthreads();
  }
  float tv = keyf(s_prefix);    // exact TOPK-th largest (reference keeps s >= tv, incl. ties)
  float mx = keyf(s_maxkey);
  float loc = 0.f;
  for (int j = tid; j < S_; j += 256) {
    float s = vals[j];
    if (s >= tv) loc += __expf(s - mx);
  }
  atomicAdd(&s_denom, loc);
  __syncthreads();
  float inv = 1.0f / s_denom;
  for (int j = tid; j < S_; j += 256) {
    float s = vals[j];
    float p = (s >= tv) ? __expf(s - mx) * inv : 0.f;
    attn[(size_t)row * S_ + j] = f2bf(p);
  }
}

// out_tile[16 x 64] = attn(bf16) @ V(bf16): each wave handles one i-tile across
// the full head dim (4 accumulators), so each A load feeds 4 WMMAs.
__global__ void k_pv(const uint16_t* __restrict__ attn, const uint16_t* __restrict__ vb,
                     uint16_t* __restrict__ ob, int b, int h) {
  int wave = threadIdx.x >> 5, lane = threadIdx.x & 31;
  int ti = blockIdx.x * 4 + wave;     // 128 i-tiles
  int i0 = ti * 16;
  const uint16_t* tileA = attn + (size_t)i0 * S_;
  const uint16_t* vt = vb + (size_t)(b * S_) * HID_ + h * HD_;
  v8f acc[4] = {};
  for (int kb = 0; kb < S_; kb += 32) {
    if (kb + 64 < S_) __builtin_prefetch((const void*)(tileA + kb + 64), 0, 0);
    v16bf a = ldA(tileA + kb, S_);
#pragma unroll
    for (int t = 0; t < 4; t++) {
      v16bf bm = ldB(vt + (size_t)kb * HID_ + t * 16, HID_);
      acc[t] = __builtin_amdgcn_wmma_f32_16x16x32_bf16(false, a, false, bm, (short)0, acc[t], false, false);
    }
  }
  int mo = (lane >> 4) * 8;
#pragma unroll
  for (int t = 0; t < 4; t++) {
    int n = h * HD_ + t * 16 + (lane & 15);
#pragma unroll
    for (int r = 0; r < 8; r++) {
      int m = b * S_ + i0 + r + mo;
      ob[(size_t)m * HID_ + n] = f2bf(acc[t][r]);
    }
  }
}

// Final projection: d_out(fp32) = O @ Wo^T. 16x64 block per wave.
__global__ void k_oproj(const uint16_t* __restrict__ ob, const uint16_t* __restrict__ wot,
                        float* __restrict__ out) {
  int wave = threadIdx.x >> 5, lane = threadIdx.x & 31;
  int n0 = (blockIdx.x * 4 + wave) * 64;
  int m0 = blockIdx.y * 16;
  const uint16_t* tileA = ob + (size_t)m0 * HID_;
  v8f acc[4] = {};
  for (int kb = 0; kb < HID_; kb += 32) {
    if (kb + 64 < HID_) __builtin_prefetch((const void*)(tileA + kb + 64), 0, 0);
    v16bf a = ldA(tileA + kb, HID_);
#pragma unroll
    for (int t = 0; t < 4; t++) {
      v16bf bm = ldB(wot + (size_t)kb * HID_ + n0 + t * 16, HID_);
      acc[t] = __builtin_amdgcn_wmma_f32_16x16x32_bf16(false, a, false, bm, (short)0, acc[t], false, false);
    }
  }
  int mo = (lane >> 4) * 8;
#pragma unroll
  for (int t = 0; t < 4; t++) {
    int n = n0 + t * 16 + (lane & 15);
#pragma unroll
    for (int r = 0; r < 8; r++) out[(size_t)(m0 + r + mo) * HID_ + n] = acc[t][r];
  }
}

extern "C" void kernel_launch(void* const* d_in, const int* in_sizes, int n_in,
                              void* d_out, int out_size, void* d_ws, size_t ws_size,
                              hipStream_t stream) {
  (void)in_sizes; (void)n_in; (void)out_size; (void)ws_size;
  const float* hs   = (const float*)d_in[0];
  const float* Wq   = (const float*)d_in[1];
  const float* Wk   = (const float*)d_in[2];
  const float* Wv   = (const float*)d_in[3];
  const float* Wo   = (const float*)d_in[4];
  const float* slpw = (const float*)d_in[5];
  const float* slpb = (const float*)d_in[6];
  const float* icpw = (const float*)d_in[7];
  const float* icpb = (const float*)d_in[8];
  const float* pw   = (const float*)d_in[9];

  char* ws = (char*)d_ws;
  size_t o = 0;
  auto take = [&](size_t bytes) { char* p = ws + o; o = (o + bytes + 255) & ~(size_t)255; return p; };
  uint16_t* hsb  = (uint16_t*)take((size_t)M_ * HID_ * 2);            // 8 MB
  uint16_t* wt   = (uint16_t*)take((size_t)4 * HID_ * HID_ * 2);      // 8 MB (q,k,v,o transposed)
  uint16_t* qb   = (uint16_t*)take((size_t)M_ * HID_ * 2);            // 8 MB
  uint16_t* ktr  = (uint16_t*)take((size_t)B_ * NH_ * HD_ * S_ * 2);  // 8 MB
  uint16_t* vb   = (uint16_t*)take((size_t)M_ * HID_ * 2);            // 8 MB
  uint16_t* ob   = (uint16_t*)take((size_t)M_ * HID_ * 2);            // 8 MB
  float*    sc   = (float*)take((size_t)S_ * S_ * 4);                 // 16 MB (reused per b,h)
  uint16_t* attn = (uint16_t*)take((size_t)S_ * S_ * 2);              // 8 MB  (reused per b,h)
  float*    meanq = (float*)take(256);
  int*      pidx  = (int*)take(256);

  k_init<<<1, 32, 0, stream>>>(meanq);
  k_convert<<<32768, 256, 0, stream>>>(Wq, Wk, Wv, Wo, hs, wt, hsb);
  k_proj_qkv<<<dim3(4, 256, 3), 128, 0, stream>>>(hsb, wt, qb, ktr, vb, meanq);
  k_pattern<<<1, 32, 0, stream>>>(meanq, slpw, slpb, icpw, icpb, pw, pidx);

  for (int bh = 0; bh < B_ * NH_; bh++) {
    int b = bh / NH_, h = bh % NH_;
    k_scores<<<dim3(8, 128), 128, 0, stream>>>(qb, ktr, pidx, sc, b, h);
    k_select_softmax<<<S_, 256, 0, stream>>>(sc, attn);
    k_pv<<<32, 128, 0, stream>>>(attn, vb, ob, b, h);
  }

  const uint16_t* wot = wt + (size_t)3 * HID_ * HID_;
  k_oproj<<<dim3(4, 256), 128, 0, stream>>>(ob, wot, (float*)d_out);
}
